// TransformerBlock_6055903887424
// MI455X (gfx1250) — matrix operand order repaired and checked by reading, not compiled, not run on hardware
//
#include <hip/hip_runtime.h>
#include <hip/hip_bf16.h>
#include <math.h>

typedef __attribute__((ext_vector_type(16))) _Float16 v16h;
typedef __attribute__((ext_vector_type(8)))  _Float16 v8h;
typedef __attribute__((ext_vector_type(8)))  float    v8f;
typedef __attribute__((ext_vector_type(4)))  float    v4f;

constexpr int BB  = 8;
constexpr int SS  = 2048;
constexpr int HH  = 256;
constexpr int NHH = 4;
constexpr int DHD = 64;
constexpr int TT  = BB * SS;        // 16384 tokens
constexpr int QKVW = 3 * HH;        // 768
constexpr int H4  = 4 * HH;         // 1024

// ---- workspace layout (bytes) ----
constexpr size_t OFF_WQKV  = 0;                               // 768*256 f16
constexpr size_t OFF_WO    = OFF_WQKV  + (size_t)QKVW*HH*2;   // 256*256 f16
constexpr size_t OFF_W1    = OFF_WO    + (size_t)HH*HH*2;     // 1024*256 f16
constexpr size_t OFF_W2    = OFF_W1    + (size_t)H4*HH*2;     // 256*1024 f16
constexpr size_t OFF_BQKV  = OFF_W2    + (size_t)HH*H4*2;     // 768 f32 (pad)
constexpr size_t OFF_XH    = OFF_BQKV  + 4096;                // T*256 f16
constexpr size_t OFF_QKVH  = OFF_XH    + (size_t)TT*HH*2;     // T*768 f16
constexpr size_t OFF_ATTNH = OFF_QKVH  + (size_t)TT*QKVW*2;   // T*256 f16
constexpr size_t OFF_H2    = OFF_ATTNH + (size_t)TT*HH*2;     // T*256 f32
constexpr size_t OFF_YH    = OFF_H2    + (size_t)TT*HH*4;     // T*256 f16
constexpr size_t OFF_GH    = OFF_YH    + (size_t)TT*HH*2;     // T*1024 f16

__device__ __forceinline__ v8f wmma_f16(v16h a, v16h b, v8f c) {
  return __builtin_amdgcn_wmma_f32_16x16x32_f16(
      /*neg_a=*/false, a, /*neg_b=*/false, b,
      /*c_mod=*/(short)0, c, /*reuse_a=*/false, /*reuse_b=*/false);
}

__device__ __forceinline__ v16h mk16(v8h lo, v8h hi) {
  v16h r;
#pragma unroll
  for (int i = 0; i < 8; ++i) { r[i] = lo[i]; r[i + 8] = hi[i]; }
  return r;
}

__device__ __forceinline__ float gelu_tanh(float x) {
  float x3 = x * x * x;
  return 0.5f * x * (1.0f + tanhf(0.7978845608028654f * (x + 0.044715f * x3)));
}

// ---------------------------------------------------------------------------
// Kernel 0: stage weights (fp32 -> f16), concat Wq|Wk|Wv and bq|bk|bv
// ---------------------------------------------------------------------------
__global__ void stage_weights(const float* __restrict__ Wq, const float* __restrict__ Wk,
                              const float* __restrict__ Wv, const float* __restrict__ Wo,
                              const float* __restrict__ W1, const float* __restrict__ W2,
                              const float* __restrict__ bq, const float* __restrict__ bk,
                              const float* __restrict__ bv,
                              _Float16* __restrict__ Wqkv_h, _Float16* __restrict__ Wo_h,
                              _Float16* __restrict__ W1_h,   _Float16* __restrict__ W2_h,
                              float* __restrict__ bqkv) {
  int i = blockIdx.x * blockDim.x + threadIdx.x;
  constexpr int N0 = QKVW * HH;          // 196608
  constexpr int N1 = N0 + HH * HH;       // +65536
  constexpr int N2 = N1 + H4 * HH;       // +262144
  constexpr int N3 = N2 + HH * H4;       // +262144
  constexpr int N4 = N3 + QKVW;          // +768
  if (i < N0) {
    int r = i / HH, c = i % HH;
    float v = (r < HH) ? Wq[r * HH + c]
            : (r < 2 * HH) ? Wk[(r - HH) * HH + c]
                           : Wv[(r - 2 * HH) * HH + c];
    Wqkv_h[i] = (_Float16)v;
  } else if (i < N1) {
    int j = i - N0; Wo_h[j] = (_Float16)Wo[j];
  } else if (i < N2) {
    int j = i - N1; W1_h[j] = (_Float16)W1[j];
  } else if (i < N3) {
    int j = i - N2; W2_h[j] = (_Float16)W2[j];
  } else if (i < N4) {
    int j = i - N3;
    bqkv[j] = (j < HH) ? bq[j] : (j < 2 * HH) ? bk[j - HH] : bv[j - 2 * HH];
  }
}

// ---------------------------------------------------------------------------
// LayerNorm: one wave (32 lanes) per row of 256; writes f16
// ---------------------------------------------------------------------------
__global__ __launch_bounds__(256) void ln_kernel(const float* __restrict__ in,
                                                 const float* __restrict__ w,
                                                 const float* __restrict__ b,
                                                 _Float16* __restrict__ out) {
  int lane = threadIdx.x & 31, wave = threadIdx.x >> 5;
  int row = blockIdx.x * 8 + wave;
  const float* p = in + (size_t)row * HH + lane * 8;
  float x[8];
  v4f a0 = *(const v4f*)p;
  v4f a1 = *(const v4f*)(p + 4);
#pragma unroll
  for (int i = 0; i < 4; ++i) { x[i] = a0[i]; x[4 + i] = a1[i]; }
  float s = 0.f, s2 = 0.f;
#pragma unroll
  for (int i = 0; i < 8; ++i) { s += x[i]; s2 += x[i] * x[i]; }
#pragma unroll
  for (int off = 16; off >= 1; off >>= 1) {
    s  += __shfl_xor(s,  off, 32);
    s2 += __shfl_xor(s2, off, 32);
  }
  float mean = s * (1.0f / HH);
  float var  = s2 * (1.0f / HH) - mean * mean;
  float inv  = rsqrtf(var + 1e-12f);
#pragma unroll
  for (int i = 0; i < 8; ++i) {
    int c = lane * 8 + i;
    out[(size_t)row * HH + c] = (_Float16)((x[i] - mean) * inv * w[c] + b[c]);
  }
}

// ---------------------------------------------------------------------------
// Generic GEMM: out[m,n] = sum_k A[m,k] * Wt[n,k] + bias[n] (+resid) (gelu?)
// One wave computes a 16x64 tile: the A fragment is loaded once per k-step and
// reused by 4 back-to-back v_wmma_f32_16x16x32_f16 (4 accumulators), giving
// the scheduler 4 independent XDL ops to overlap with the next clause of
// global_load_b128s. Weights/activations are L2-resident; HBM touched once.
// ---------------------------------------------------------------------------
template <int K, bool RESID, bool GELU_, bool W32, bool W16>
__global__ __launch_bounds__(256) void gemm_kernel(const _Float16* __restrict__ A,
                                                   const _Float16* __restrict__ Wt,
                                                   const float* __restrict__ bias,
                                                   const float* __restrict__ resid,
                                                   float* __restrict__ out32,
                                                   _Float16* __restrict__ out16,
                                                   int N) {
  int lane = threadIdx.x & 31, wave = threadIdx.x >> 5;
  int m0 = (blockIdx.y * 8 + wave) * 16;
  int n0 = blockIdx.x * 64;
  int l15 = lane & 15;
  bool hiL = (lane & 16) != 0;
  const _Float16* arow  = A  + (size_t)(m0 + l15) * K;
  const _Float16* brow0 = Wt + (size_t)(n0 + l15) * K;
  int ak = hiL ? 8 : 0;    // A: lane-half K split {0..7,16..23} / {8..15,24..31}
  int bk = hiL ? 8 : 0;    // B: lane-half K split {0..7,16..23} / {8..15,24..31}
  v8f acc[4] = {{}, {}, {}, {}};
#pragma unroll 4
  for (int k0 = 0; k0 < K; k0 += 32) {
    __builtin_prefetch(arow + k0 + 128, 0, 1);   // global_prefetch_b8 on A stream
    v8h alo = *(const v8h*)(arow + k0 + ak);
    v8h ahi = *(const v8h*)(arow + k0 + ak + 16);
    v16h af = mk16(alo, ahi);
#pragma unroll
    for (int nb = 0; nb < 4; ++nb) {
      const _Float16* br = brow0 + (size_t)nb * 16 * K + k0 + bk;
      acc[nb] = wmma_f16(af, mk16(*(const v8h*)br, *(const v8h*)(br + 16)), acc[nb]);
    }
  }
  int mbase = m0 + (hiL ? 8 : 0);
#pragma unroll
  for (int nb = 0; nb < 4; ++nb) {
    int n = n0 + nb * 16 + l15;
    float bn = bias[n];
#pragma unroll
    for (int r = 0; r < 8; ++r) {
      int m = mbase + r;
      float v = acc[nb][r] + bn;
      if (RESID) v += resid[(size_t)m * N + n];
      if (GELU_) v = gelu_tanh(v);
      if (W32) out32[(size_t)m * N + n] = v;
      if (W16) out16[(size_t)m * N + n] = (_Float16)v;
    }
  }
}

// ---------------------------------------------------------------------------
// Flash attention: one wave handles 16 queries of one (b,h).
// Scores computed transposed (S^T = K·Q^T) so each query owns one lane column;
// online softmax with xor-16 shuffle; O^T += V^T · P^T via LDS staging.
// ---------------------------------------------------------------------------
__global__ __launch_bounds__(128) void attn_kernel(const _Float16* __restrict__ qkv,
                                                   const int* __restrict__ mask,
                                                   _Float16* __restrict__ attn_out) {
  __shared__ __align__(16) _Float16 sm[4][64 * 32 + 16 * 32]; // per-wave V^T + P
  int lane = threadIdx.x & 31, wave = threadIdx.x >> 5;
  _Float16* vT   = sm[wave];
  _Float16* pbuf = sm[wave] + 64 * 32;

  int gid = blockIdx.x * 4 + wave;         // B*NH*(S/16) = 4096 waves
  int qt = gid & 127;                      // S/16 = 128
  int h  = (gid >> 7) & 3;
  int b  = gid >> 9;
  int s0q = qt * 16;
  int l15 = lane & 15;
  bool hiL = (lane & 16) != 0;
  size_t tokbase = (size_t)b * SS;

  // Q fragments for the B-operand (B[d][q]): loaded once
  const _Float16* qrow = qkv + (tokbase + s0q + l15) * QKVW + h * DHD;
  int bkoff = hiL ? 8 : 0;
  v16h bq0 = mk16(*(const v8h*)(qrow + 0  + bkoff), *(const v8h*)(qrow + 0  + bkoff + 16));
  v16h bq1 = mk16(*(const v8h*)(qrow + 32 + bkoff), *(const v8h*)(qrow + 32 + bkoff + 16));
  int mrow = mask[b * SS + s0q + l15];     // query-row mask

  float m_run = -3.0e38f, l_run = 0.0f;
  v8f oacc[4] = {{}, {}, {}, {}};
  int akoff = hiL ? 8 : 0;

  for (int kk = 0; kk < SS; kk += 32) {
    // ---- stage V^T (32 keys x 64 d -> LDS [d][key]) ----
    const _Float16* vrow = qkv + (tokbase + kk + lane) * QKVW + 2 * HH + h * DHD;
    _Float16 vv[64];
#pragma unroll
    for (int j = 0; j < 8; ++j) {
      v8h t = *(const v8h*)(vrow + j * 8);
#pragma unroll
      for (int e = 0; e < 8; ++e) vv[j * 8 + e] = t[e];
    }
#pragma unroll
    for (int j = 0; j < 64; ++j) vT[j * 32 + lane] = vv[j];

    // ---- scores (two 16-key tiles, transposed): S^T = K · Q^T ----
    const _Float16* k0r = qkv + (tokbase + kk +      l15) * QKVW + HH + h * DHD;
    const _Float16* k1r = qkv + (tokbase + kk + 16 + l15) * QKVW + HH + h * DHD;
    v8f st0 = {}, st1 = {};
    {
      v16h a00 = mk16(*(const v8h*)(k0r + 0  + akoff), *(const v8h*)(k0r + 0  + akoff + 16));
      v16h a01 = mk16(*(const v8h*)(k0r + 32 + akoff), *(const v8h*)(k0r + 32 + akoff + 16));
      st0 = wmma_f16(a00, bq0, st0);
      st0 = wmma_f16(a01, bq1, st0);
      v16h a10 = mk16(*(const v8h*)(k1r + 0  + akoff), *(const v8h*)(k1r + 0  + akoff + 16));
      v16h a11 = mk16(*(const v8h*)(k1r + 32 + akoff), *(const v8h*)(k1r + 32 + akoff + 16));
      st1 = wmma_f16(a10, bq0, st1);
      st1 = wmma_f16(a11, bq1, st1);
    }

    // ---- online softmax over these 32 keys (per query = per lane column) ----
    float sv[16];
#pragma unroll
    for (int r = 0; r < 8; ++r) {
      sv[r]     = mrow ? st0[r] * 0.125f : -1e9f;   // 1/sqrt(64)
      sv[8 + r] = mrow ? st1[r] * 0.125f : -1e9f;
    }
    float mx = sv[0];
#pragma unroll
    for (int r = 1; r < 16; ++r) mx = fmaxf(mx, sv[r]);
    mx = fmaxf(mx, __shfl_xor(mx, 16, 32));
    float m_new = fmaxf(m_run, mx);
    float alpha = expf(m_run - m_new);
    float lsum = 0.0f;
#pragma unroll
    for (int r = 0; r < 16; ++r) { float p = expf(sv[r] - m_new); sv[r] = p; lsum += p; }
    lsum += __shfl_xor(lsum, 16, 32);
    l_run = l_run * alpha + lsum;
#pragma unroll
    for (int d = 0; d < 4; ++d) oacc[d] *= alpha;

    // ---- P^T -> LDS as [query][key] f16 ----
    int kb = hiL ? 8 : 0;
#pragma unroll
    for (int r = 0; r < 8; ++r) {
      pbuf[l15 * 32 + kb + r]      = (_Float16)sv[r];
      pbuf[l15 * 32 + 16 + kb + r] = (_Float16)sv[8 + r];
    }
    asm volatile("s_wait_dscnt 0x0" ::: "memory");   // LDS in-order + fence reorder

    // ---- O^T += V^T · P^T  (4 d-blocks of 16) ----
    v16h pB = mk16(*(const v8h*)(pbuf + l15 * 32 + bkoff),
                   *(const v8h*)(pbuf + l15 * 32 + bkoff + 16));
#pragma unroll
    for (int dblk = 0; dblk < 4; ++dblk) {
      const _Float16* vr = vT + (dblk * 16 + l15) * 32;
      v16h vA = mk16(*(const v8h*)(vr + akoff), *(const v8h*)(vr + akoff + 16));
      oacc[dblk] = wmma_f16(vA, pB, oacc[dblk]);
    }
    m_run = m_new;
  }

  // ---- writeback: O[q][d] = O^T / l  (f16 for the Wo GEMM) ----
  float invl = 1.0f / l_run;
  int q = s0q + l15;
  _Float16* orow = attn_out + (tokbase + q) * HH + h * DHD;
  int dbase = hiL ? 8 : 0;
#pragma unroll
  for (int dblk = 0; dblk < 4; ++dblk)
#pragma unroll
    for (int r = 0; r < 8; ++r)
      orow[dblk * 16 + dbase + r] = (_Float16)(oacc[dblk][r] * invl);
}

// ---------------------------------------------------------------------------
extern "C" void kernel_launch(void* const* d_in, const int* in_sizes, int n_in,
                              void* d_out, int out_size, void* d_ws, size_t ws_size,
                              hipStream_t stream) {
  const float* hidden = (const float*)d_in[0];
  const float* Wq = (const float*)d_in[1];  const float* bq = (const float*)d_in[2];
  const float* Wk = (const float*)d_in[3];  const float* bk = (const float*)d_in[4];
  const float* Wv = (const float*)d_in[5];  const float* bv = (const float*)d_in[6];
  const float* Wo = (const float*)d_in[7];  const float* bo = (const float*)d_in[8];
  const float* W1 = (const float*)d_in[9];  const float* b1 = (const float*)d_in[10];
  const float* W2 = (const float*)d_in[11]; const float* b2 = (const float*)d_in[12];
  const float* ln1_w = (const float*)d_in[13]; const float* ln1_b = (const float*)d_in[14];
  const float* ln2_w = (const float*)d_in[15]; const float* ln2_b = (const float*)d_in[16];
  const int*   mask  = (const int*)d_in[17];
  float* out = (float*)d_out;

  char* ws = (char*)d_ws;
  _Float16* Wqkv_h = (_Float16*)(ws + OFF_WQKV);
  _Float16* Wo_h   = (_Float16*)(ws + OFF_WO);
  _Float16* W1_h   = (_Float16*)(ws + OFF_W1);
  _Float16* W2_h   = (_Float16*)(ws + OFF_W2);
  float*    bqkv   = (float*)   (ws + OFF_BQKV);
  _Float16* x_h    = (_Float16*)(ws + OFF_XH);
  _Float16* qkv_h  = (_Float16*)(ws + OFF_QKVH);
  _Float16* attn_h = (_Float16*)(ws + OFF_ATTNH);
  float*    h2     = (float*)   (ws + OFF_H2);
  _Float16* y_h    = (_Float16*)(ws + OFF_YH);
  _Float16* g_h    = (_Float16*)(ws + OFF_GH);

  // 0) weights -> f16
  {
    int tot = QKVW * HH + HH * HH + H4 * HH + HH * H4 + QKVW;
    stage_weights<<<(tot + 255) / 256, 256, 0, stream>>>(
        Wq, Wk, Wv, Wo, W1, W2, bq, bk, bv, Wqkv_h, Wo_h, W1_h, W2_h, bqkv);
  }
  // 1) x = LN1(hidden) -> f16
  ln_kernel<<<TT / 8, 256, 0, stream>>>(hidden, ln1_w, ln1_b, x_h);
  // 2) qkv = x @ Wqkv^T + bqkv  (T x 768)
  gemm_kernel<HH, false, false, false, true>
      <<<dim3(QKVW / 64, TT / 128), 256, 0, stream>>>(
          x_h, Wqkv_h, bqkv, nullptr, nullptr, qkv_h, QKVW);
  // 3) flash attention -> attn_h (f16)
  attn_kernel<<<(BB * NHH * (SS / 16)) / 4, 128, 0, stream>>>(qkv_h, mask, attn_h);
  // 4) h2 = hidden + attn @ Wo^T + bo
  gemm_kernel<HH, true, false, true, false>
      <<<dim3(HH / 64, TT / 128), 256, 0, stream>>>(
          attn_h, Wo_h, bo, hidden, h2, nullptr, HH);
  // 5) y = LN2(h2) -> f16
  ln_kernel<<<TT / 8, 256, 0, stream>>>(h2, ln2_w, ln2_b, y_h);
  // 6) g = gelu(y @ W1^T + b1) -> f16 (T x 1024)
  gemm_kernel<HH, false, true, false, true>
      <<<dim3(H4 / 64, TT / 128), 256, 0, stream>>>(
          y_h, W1_h, b1, nullptr, nullptr, g_h, H4);
  // 7) out = h2 + g @ W2^T + b2
  gemm_kernel<H4, true, false, true, false>
      <<<dim3(HH / 64, TT / 128), 256, 0, stream>>>(
          g_h, W2_h, b2, h2, out, nullptr, HH);
}
